// RNNWrapper_57432302682695
// MI455X (gfx1250) — matrix operand order repaired and checked by reading, not compiled, not run on hardware
//
#include <hip/hip_runtime.h>
#include <hip/hip_bf16.h>

// Problem constants (match reference): B=128, T=1024, V=32000, E=512, H=512, O=5
#define B_   128
#define T_   1024
#define E_   512
#define H_   512
#define O_   5
#define KTOT 1024   // E + H (concat order: [h, x])

typedef __attribute__((ext_vector_type(16))) __bf16 v16bf;
typedef __attribute__((ext_vector_type(8)))  __bf16 v8bf;
typedef __attribute__((ext_vector_type(8)))  float  v8f;

static __device__ __forceinline__ v16bf cat8(v8bf lo, v8bf hi) {
  return __builtin_shufflevector(lo, hi, 0,1,2,3,4,5,6,7,8,9,10,11,12,13,14,15);
}

static __device__ __forceinline__ v8f wmma_bf16(v16bf a, v16bf b, v8f c) {
  // D = A(16x32 bf16) * B(32x16 bf16) + C(16x16 f32)
  return __builtin_amdgcn_wmma_f32_16x16x32_bf16(
      /*neg_a=*/false, a, /*neg_b=*/false, b,
      /*c_mod=*/(short)0, c, /*reuse_a=*/false, /*reuse_b=*/false);
}

static __device__ __forceinline__ float sigmoidf_(float x) {
  return 1.0f / (1.0f + __expf(-x));
}

// ---------------------------------------------------------------------------
// Pack fused gate weights into WMMA B-matrix tile layout (bf16).
// Layout: Wp[b_blk(32)][gate(4)][kt(32)][lane(32)][16 bf16]
//   lane L of a 32x16 B tile holds column n = L&15, K = (L>>4)*8 .. +7 then 16+(L>>4)*8 .. +7
//   (16 bf16 = one 32-byte per-lane load in the hot loop)
// K index: 0..511 -> h rows of W, 512..1023 -> x rows  (concat([h,x]) @ W)
// ---------------------------------------------------------------------------
__global__ __launch_bounds__(256) void pack_weights_kernel(
    const float* __restrict__ wf, const float* __restrict__ wi,
    const float* __restrict__ wo, const float* __restrict__ wc,
    __bf16* __restrict__ Wp)
{
  const int gid = blockIdx.x * blockDim.x + threadIdx.x;  // 32*4*32*32 = 131072
  if (gid >= 32 * 4 * 32 * 32) return;
  const int lane = gid & 31;
  const int kt   = (gid >> 5) & 31;
  const int g    = (gid >> 10) & 3;
  const int bblk = gid >> 12;
  const float* w = (g == 0) ? wf : (g == 1) ? wi : (g == 2) ? wo : wc;  // [1024,512] row-major
  const int n     = bblk * 16 + (lane & 15);
  const int kbase = kt * 32 + (lane >> 4) * 8;
  v16bf out;
#pragma unroll
  for (int j = 0; j < 16; ++j) out[j] = (__bf16)w[(size_t)(kbase + (j < 8 ? j : j + 8)) * H_ + n];
  *(v16bf*)(Wp + (size_t)gid * 16) = out;
}

__global__ void zero_kernel(unsigned int* __restrict__ p, int n) {
  const int i = blockIdx.x * blockDim.x + threadIdx.x;
  if (i < n) p[i] = 0u;
}

// --- tile loaders for the pipelined K loop ---------------------------------
static __device__ __forceinline__ void load_tiles_h(
    int kt, const __bf16* __restrict__ hrow, const __bf16* __restrict__ wbase,
    int lane, int khalf,
    v16bf& a, v16bf& bF, v16bf& bI, v16bf& bO, v16bf& bC)
{
  const __bf16* ap = hrow + kt * 32 + khalf * 8;
  a = cat8(*(const v8bf*)ap, *(const v8bf*)(ap + 16));
  const __bf16* wp0 = wbase + ((size_t)kt * 32 + lane) * 16;
  bF = *(const v16bf*)(wp0);
  bI = *(const v16bf*)(wp0 + 16384);
  bO = *(const v16bf*)(wp0 + 32768);
  bC = *(const v16bf*)(wp0 + 49152);
}

static __device__ __forceinline__ void load_tiles_x(
    int kt, const float* __restrict__ xrow, const __bf16* __restrict__ wbase,
    int lane, int khalf,
    v16bf& a, v16bf& bF, v16bf& bI, v16bf& bO, v16bf& bC)
{
  const float* xp = xrow + (kt - 16) * 32 + khalf * 8;
  v8bf c0, c1;
#pragma unroll
  for (int j = 0; j < 8; ++j) {
    c0[j] = (__bf16)xp[j];
    c1[j] = (__bf16)xp[j + 16];
  }
  a = cat8(c0, c1);
  const __bf16* wp0 = wbase + ((size_t)kt * 32 + lane) * 16;
  bF = *(const v16bf*)(wp0);
  bI = *(const v16bf*)(wp0 + 16384);
  bO = *(const v16bf*)(wp0 + 32768);
  bC = *(const v16bf*)(wp0 + 49152);
}

// ---------------------------------------------------------------------------
// One LSTM timestep. Grid = 32 WGs (each owns 16 h-columns), 256 thr = 8 waves.
// Wave w computes the [16 x 16] z-tiles (f,i,o,c) for batch rows 16w..16w+15.
// K loop: kt 0..15 reads h_in (bf16), kt 16..31 gathers embedding (fp32->bf16).
// Double-buffered: loads for kt+1 are issued before the WMMAs of kt so waits
// are partial (loads in flight across the XDL work), not full round-trips.
// ---------------------------------------------------------------------------
__global__ __launch_bounds__(256) void lstm_step_kernel(
    const int*   __restrict__ inputs, const int* __restrict__ seq_lengths,
    const float* __restrict__ embed,
    const float* __restrict__ b_f, const float* __restrict__ b_i,
    const float* __restrict__ b_o, const float* __restrict__ b_c,
    const __bf16* __restrict__ Wp,
    float* __restrict__ Cst, const __bf16* __restrict__ h_in,
    __bf16* __restrict__ h_out, float* __restrict__ lastH, int t)
{
  const int lane  = threadIdx.x & 31;
  const int wave  = threadIdx.x >> 5;   // 0..7 -> M tile
  const int nblk  = blockIdx.x;         // 0..31 -> h-column block
  const int mloc  = lane & 15;
  const int khalf = lane >> 4;          // 0/1: which K-half this lane holds
  const int mrow  = wave * 16 + mloc;   // batch row whose A data this lane loads

  // ---- hoisted scalar/state loads: latency hides under the GEMM ----
  const int   ncol = nblk * 16 + mloc;
  const float bFv = b_f[ncol], bIv = b_i[ncol], bOv = b_o[ncol], bCv = b_c[ncol];
  float cOld[8];
  int   slen[8];
#pragma unroll
  for (int r = 0; r < 8; ++r) {
    const int row = wave * 16 + khalf * 8 + r;
    cOld[r] = Cst[(size_t)row * H_ + ncol];
    slen[r] = seq_lengths[row];
  }

  const int token = inputs[(size_t)mrow * T_ + t];
  const float*  xrow = embed + (size_t)token * E_;
  const __bf16* hrow = h_in + (size_t)mrow * H_;

  v8f accF = {}; v8f accI = {}; v8f accO = {}; v8f accC = {};

  // This block's packed weights: stride between gates = 32*32*16 = 16384 elems
  const __bf16* wbase = Wp + (size_t)nblk * 4 * 32 * 32 * 16;

  // ---- software-pipelined K loop (double buffered) ----
  v16bf a, bF, bI, bO, bC;       // current tiles
  v16bf an, bFn, bIn, bOn, bCn;  // next tiles

  load_tiles_h(0, hrow, wbase, lane, khalf, a, bF, bI, bO, bC);

  // h part: kt = 0..15 (prefetch kt+1)
#pragma unroll
  for (int kt = 0; kt < 15; ++kt) {
    load_tiles_h(kt + 1, hrow, wbase, lane, khalf, an, bFn, bIn, bOn, bCn);
    accF = wmma_bf16(a, bF, accF);
    accI = wmma_bf16(a, bI, accI);
    accO = wmma_bf16(a, bO, accO);
    accC = wmma_bf16(a, bC, accC);
    a = an; bF = bFn; bI = bIn; bO = bOn; bC = bCn;
  }
  // kt = 15: prefetch first x tile
  load_tiles_x(16, xrow, wbase, lane, khalf, an, bFn, bIn, bOn, bCn);
  accF = wmma_bf16(a, bF, accF);
  accI = wmma_bf16(a, bI, accI);
  accO = wmma_bf16(a, bO, accO);
  accC = wmma_bf16(a, bC, accC);
  a = an; bF = bFn; bI = bIn; bO = bOn; bC = bCn;

  // x part: kt = 16..31 (prefetch kt+1)
#pragma unroll
  for (int kt = 16; kt < 31; ++kt) {
    load_tiles_x(kt + 1, xrow, wbase, lane, khalf, an, bFn, bIn, bOn, bCn);
    accF = wmma_bf16(a, bF, accF);
    accI = wmma_bf16(a, bI, accI);
    accO = wmma_bf16(a, bO, accO);
    accC = wmma_bf16(a, bC, accC);
    a = an; bF = bFn; bI = bIn; bO = bOn; bC = bCn;
  }
  // kt = 31 (drain)
  accF = wmma_bf16(a, bF, accF);
  accI = wmma_bf16(a, bI, accI);
  accO = wmma_bf16(a, bO, accO);
  accC = wmma_bf16(a, bC, accC);

  // ---- gates + state update ----
  // C/D layout: lane L, VGPR r -> row m = r + 8*(L>>4), col n = L&15
#pragma unroll
  for (int r = 0; r < 8; ++r) {
    const int row = wave * 16 + khalf * 8 + r;
    const float f  = sigmoidf_(accF[r] + bFv);
    const float i  = sigmoidf_(accI[r] + bIv);
    const float o  = sigmoidf_(accO[r] + bOv);
    const float ch = tanhf(accC[r] + bCv);
    const size_t idx = (size_t)row * H_ + ncol;
    const float cN = f * cOld[r] + i * ch;
    Cst[idx] = cN;
    const float h = o * tanhf(cN);
    h_out[idx] = (__bf16)h;
    if (slen[r] == t + 1) lastH[idx] = h;  // h at t = seq_len-1
  }
}

// ---------------------------------------------------------------------------
// Final FC: out[m][o] = lastH[m,:] . fc_w[:,o] + fc_b[o]   (128 x 5, tiny)
// ---------------------------------------------------------------------------
__global__ __launch_bounds__(256) void fc_kernel(
    const float* __restrict__ lastH, const float* __restrict__ fcw,
    const float* __restrict__ fcb, float* __restrict__ out)
{
  const int m = blockIdx.x;
  const int wave = threadIdx.x >> 5;
  const int lane = threadIdx.x & 31;
  if (wave >= O_) return;
  float s = 0.0f;
  for (int k = lane; k < H_; k += 32)
    s += lastH[(size_t)m * H_ + k] * fcw[(size_t)k * O_ + wave];
#pragma unroll
  for (int off = 16; off > 0; off >>= 1) s += __shfl_down(s, off, 32);
  if (lane == 0) out[m * O_ + wave] = s + fcb[wave];
}

// ---------------------------------------------------------------------------
extern "C" void kernel_launch(void* const* d_in, const int* in_sizes, int n_in,
                              void* d_out, int out_size, void* d_ws, size_t ws_size,
                              hipStream_t stream) {
  const int*   inputs = (const int*)  d_in[0];
  const int*   seqlen = (const int*)  d_in[1];
  const float* embed  = (const float*)d_in[2];
  const float* w_f    = (const float*)d_in[3];
  const float* b_f    = (const float*)d_in[4];
  const float* w_i    = (const float*)d_in[5];
  const float* b_i    = (const float*)d_in[6];
  const float* w_o    = (const float*)d_in[7];
  const float* b_o    = (const float*)d_in[8];
  const float* w_c    = (const float*)d_in[9];
  const float* b_c    = (const float*)d_in[10];
  const float* fcw    = (const float*)d_in[11];
  const float* fcb    = (const float*)d_in[12];
  float* out = (float*)d_out;

  // Workspace layout (total ~4.75 MB):
  //   [0, 4MB)          packed bf16 gate weights
  //   [4MB, +256KB)     cell state C (fp32)
  //   [+256KB, +512KB)  lastH (fp32)
  //   [+512KB, +640KB)  h ping (bf16)
  //   [+640KB, +768KB)  h pong (bf16)
  char* ws = (char*)d_ws;
  __bf16* Wp    = (__bf16*)ws;
  float*  Cst   = (float*) (ws + (4u << 20));
  float*  lastH = (float*) (ws + (4u << 20) + (256u << 10));
  __bf16* h0    = (__bf16*)(ws + (4u << 20) + (512u << 10));
  __bf16* h1    = (__bf16*)(ws + (4u << 20) + (640u << 10));

  // Phase 0: pack weights to WMMA tile layout; zero state/h/lastH (768 KB).
  pack_weights_kernel<<<512, 256, 0, stream>>>(w_f, w_i, w_o, w_c, Wp);
  zero_kernel<<<(196608 + 255) / 256, 256, 0, stream>>>(
      (unsigned int*)(ws + (4u << 20)), 196608);

  // Phase 1: 1024 sequential LSTM steps (kernel boundaries = grid-wide sync,
  // ping-pong h buffers so no block reads another block's same-step write).
  __bf16* hin = h0;
  __bf16* hout = h1;
  for (int t = 0; t < T_; ++t) {
    lstm_step_kernel<<<32, 256, 0, stream>>>(inputs, seqlen, embed,
                                             b_f, b_i, b_o, b_c,
                                             Wp, Cst, hin, hout, lastH, t);
    __bf16* tmp = hin; hin = hout; hout = tmp;
  }

  // Phase 2: final projection [128,512] x [512,5] + bias.
  fc_kernel<<<128, 256, 0, stream>>>(lastH, fcw, fcb, out);
}
